// InceptionResidualBlock_28913719837039
// MI455X (gfx1250) — matrix operand order repaired and checked by reading, not compiled, not run on hardware
//
#include <hip/hip_runtime.h>
#include <hip/hip_bf16.h>
#include <stdint.h>

typedef __attribute__((ext_vector_type(16))) __bf16 v16bf;
typedef __attribute__((ext_vector_type(8)))  float  v8f;

struct U128x2 { uint4 a, b; };
union Frag { U128x2 u; v16bf v; };

__device__ __forceinline__ unsigned short f2bf(float f) {
  unsigned int u = __float_as_uint(f);
  u += 0x7FFFu + ((u >> 16) & 1u);          // round-to-nearest-even
  return (unsigned short)(u >> 16);
}

// Low 32 bits of a generic pointer to LDS == LDS byte address.
__device__ __forceinline__ unsigned lds_addr32(const void* p) {
  return (unsigned)(uintptr_t)p;
}

// ---------------- fp32 -> bf16 bulk convert (8 elems/thread) ----------------
__global__ void cvt_bf16_kernel(const float* __restrict__ x,
                                unsigned short* __restrict__ y, long long n) {
  long long i = ((long long)blockIdx.x * blockDim.x + threadIdx.x) * 8;
  if (i >= n) return;
  const float4* p = (const float4*)(x + i);
  float4 a = p[0], b = p[1];
  uint4 o;
  o.x = f2bf(a.x) | ((unsigned)f2bf(a.y) << 16);
  o.y = f2bf(a.z) | ((unsigned)f2bf(a.w) << 16);
  o.z = f2bf(b.x) | ((unsigned)f2bf(b.y) << 16);
  o.w = f2bf(b.z) | ((unsigned)f2bf(b.w) << 16);
  *(uint4*)(y + i) = o;
}

// ------- pack W[k][K][N] (fp32) into WMMA-B bf16 fragment layout -----------
// Fragment element e of lane l for (k, chunk c, ntile t):
//   K = 32c + (l>=16 ? 16 : 0) + e,  N = (l&15) + 16t
__global__ void pack_w_kernel(const float* __restrict__ W,
                              unsigned short* __restrict__ Wp,
                              int Kk, int Cin, int Cout) {
  int idx = blockIdx.x * blockDim.x + threadIdx.x;
  int NT = Cout >> 4, CC = Cin >> 5;
  int total = Kk * CC * NT * 32;
  if (idx >= total) return;
  int lane = idx & 31;
  int g = idx >> 5;
  int t = g % NT; g /= NT;
  int c = g % CC; g /= CC;
  int k = g;
  int n  = (lane & 15) + (t << 4);
  int kb = (c << 5) + ((lane >= 16) ? 16 : 0);
  unsigned short* d = Wp + (size_t)idx * 16;
  for (int e = 0; e < 16; ++e)
    d[e] = f2bf(W[((size_t)(k * Cin + kb + e)) * Cout + n]);
}

// ---------------- gathered GEMM: out = act(sum_k gather_k(in) @ W_k + b) ----
// OUTMODE 0: bf16 output [n, COUT], ReLU
// OUTMODE 1: fp32 output to d_out slice (stride 256), ReLU then + residual x
template <int CIN, int COUT, int KK, int OUTMODE>
__global__ void __launch_bounds__(128) gconv_kernel(
    const unsigned short* __restrict__ in, const int* __restrict__ nbr,
    const unsigned short* __restrict__ Wp, const float* __restrict__ bias,
    void* __restrict__ outp, const float* __restrict__ xres, int n) {
  constexpr int CC = CIN / 32;   // K-chunks of 32
  constexpr int NT = COUT / 16;  // N-tiles of 16
  extern __shared__ unsigned short smw[];  // CIN*COUT bf16 = one W_k

  const int lane = threadIdx.x & 31;
  const int wave = threadIdx.x >> 5;
  const int tile = blockIdx.x * 4 + wave;  // 16-row tile per wave
  const int m0 = tile << 4;
  int pt = m0 + (lane & 15);
  int ptc = pt < n ? pt : (n - 1);         // clamp so OOB waves stay safe
  const int aoff = (lane >= 16) ? 16 : 0;  // A-fragment lane-half K offset

  const unsigned lds_base = lds_addr32(smw);

  v8f zero = {0.f, 0.f, 0.f, 0.f, 0.f, 0.f, 0.f, 0.f};
  v8f acc[NT];
  for (int t = 0; t < NT; ++t) acc[t] = zero;

  for (int k = 0; k < KK; ++k) {
    __syncthreads();  // previous k's LDS reads done
    {   // stage W_k into LDS via the async Global->LDS engine (ASYNCcnt)
      const uint4* src = (const uint4*)(Wp + (size_t)k * (CIN * COUT));
      for (int i = threadIdx.x; i < (CIN * COUT) / 8; i += 128) {
        unsigned ldsa = lds_base + (unsigned)i * 16u;
        const uint4* ga = src + i;
        asm volatile("global_load_async_to_lds_b128 %0, %1, off"
                     :: "v"(ldsa), "v"(ga)
                     : "memory");
      }
      asm volatile("s_wait_asynccnt 0x0" ::: "memory");
    }
    if (k + 1 < KK)
      __builtin_prefetch(Wp + (size_t)(k + 1) * (CIN * COUT), 0, 0);
    __syncthreads();  // all waves' async stages committed to LDS

    int ri = (KK > 1) ? nbr[(size_t)ptc * KK + k] : ptc;
    const unsigned short* rowp = in + (size_t)ri * CIN + aoff;

    for (int c = 0; c < CC; ++c) {
      Frag af;  // A: two contiguous 16B chunks of the gathered bf16 row
      const uint4* ap = (const uint4*)(rowp + 32 * c);
      af.u.a = ap[0];   // K = base + 0..7
      af.u.b = ap[1];   // K = base + 8..15
      for (int t = 0; t < NT; ++t) {
        Frag bfrag;
        const uint4* bp =
            (const uint4*)(smw + (size_t)(((c * NT + t) * 32 + lane) * 16));
        bfrag.u.a = bp[0];
        bfrag.u.b = bp[1];
        acc[t] = __builtin_amdgcn_wmma_f32_16x16x32_bf16(
            false, af.v, false, bfrag.v, (short)0, acc[t], false, false);
      }
    }
  }

  // epilogue: C/D layout -> row = m0 + r + (lane>=16?8:0), col = (lane&15)+16t
  const int mb = m0 + ((lane >= 16) ? 8 : 0);
  const int c0 = lane & 15;
  for (int t = 0; t < NT; ++t) {
    const int col = c0 + (t << 4);
    const float bv = bias[col];
    for (int r = 0; r < 8; ++r) {
      const int row = mb + r;
      if (row < n) {
        float v = acc[t][r] + bv;
        v = v > 0.f ? v : 0.f;
        if (OUTMODE == 0) {
          ((unsigned short*)outp)[(size_t)row * COUT + col] = f2bf(v);
        } else {
          ((float*)outp)[(size_t)row * 256 + col] =
              v + xres[(size_t)row * 256 + col];
        }
      }
    }
  }
}

extern "C" void kernel_launch(void* const* d_in, const int* in_sizes, int n_in,
                              void* d_out, int out_size, void* d_ws,
                              size_t ws_size, hipStream_t stream) {
  const float* x   = (const float*)d_in[0];
  const int*   nbr = (const int*)d_in[1];
  const float* W20 = (const float*)d_in[2];
  const float* b20 = (const float*)d_in[3];
  const float* W21 = (const float*)d_in[4];
  const float* b21 = (const float*)d_in[5];
  const float* W10 = (const float*)d_in[6];
  const float* b10 = (const float*)d_in[7];
  const float* W11 = (const float*)d_in[8];
  const float* b11 = (const float*)d_in[9];
  const float* W12 = (const float*)d_in[10];
  const float* b12 = (const float*)d_in[11];
  float* out = (float*)d_out;

  const int C = 256;
  const int N = in_sizes[0] / C;
  const int K = 27;

  char* ws = (char*)d_ws;
  size_t off = 0;
  auto take = [&](size_t bytes) {
    char* p = ws + off;
    off = (off + bytes + 255) & ~(size_t)255;
    return p;
  };
  unsigned short* xbf = (unsigned short*)take((size_t)N * 256 * 2);
  unsigned short* h20 = (unsigned short*)take((size_t)N * 64 * 2);
  unsigned short* h10 = (unsigned short*)take((size_t)N * 64 * 2);
  unsigned short* h11 = (unsigned short*)take((size_t)N * 64 * 2);
  unsigned short* p20 = (unsigned short*)take((size_t)K * 256 * 64 * 2);
  unsigned short* p21 = (unsigned short*)take((size_t)K * 64 * 128 * 2);
  unsigned short* p10 = (unsigned short*)take((size_t)256 * 64 * 2);
  unsigned short* p11 = (unsigned short*)take((size_t)K * 64 * 64 * 2);
  unsigned short* p12 = (unsigned short*)take((size_t)64 * 128 * 2);

  long long nelem = (long long)N * 256;
  cvt_bf16_kernel<<<(unsigned)((nelem / 8 + 255) / 256), 256, 0, stream>>>(
      x, xbf, nelem);

  auto pack = [&](const float* W, unsigned short* Wp, int Kk, int Cin,
                  int Cout) {
    int total = Kk * (Cin / 32) * (Cout / 16) * 32;
    pack_w_kernel<<<(total + 127) / 128, 128, 0, stream>>>(W, Wp, Kk, Cin,
                                                           Cout);
  };
  pack(W20, p20, K, 256, 64);
  pack(W21, p21, K, 64, 128);
  pack(W10, p10, 1, 256, 64);
  pack(W11, p11, K, 64, 64);
  pack(W12, p12, 1, 64, 128);

  int tiles = (N + 15) / 16;
  unsigned grid = (unsigned)((tiles + 3) / 4);

  // chain "1" (literal code): h20 = relu(conv20(x)); chain1 = relu(conv21(h20))
  gconv_kernel<256, 64, 27, 0>
      <<<grid, 128, 256 * 64 * 2, stream>>>(xbf, nbr, p20, b20, h20, nullptr, N);
  gconv_kernel<64, 128, 27, 1>
      <<<grid, 128, 64 * 128 * 2, stream>>>(h20, nbr, p21, b21, out + 128,
                                            x + 128, N);
  // chain "2": h10 = relu(x@W10); h11 = relu(conv11(h10)); chain2 = relu(h11@W12)
  gconv_kernel<256, 64, 1, 0>
      <<<grid, 128, 256 * 64 * 2, stream>>>(xbf, nullptr, p10, b10, h10,
                                            nullptr, N);
  gconv_kernel<64, 64, 27, 0>
      <<<grid, 128, 64 * 64 * 2, stream>>>(h10, nbr, p11, b11, h11, nullptr, N);
  gconv_kernel<64, 128, 1, 1>
      <<<grid, 128, 64 * 128 * 2, stream>>>(h11, nullptr, p12, b12, out,
                                            x, N);

  (void)n_in; (void)out_size; (void)ws_size;
}
